// SkipLSTMCell_18416819765876
// MI455X (gfx1250) — compile-verified, matrix operand order checked by reading, not run on hardware
//
#include <hip/hip_runtime.h>
#include <hip/hip_bf16.h>
#include <math.h>

// SkipLSTM cell, B=262144, IC=HC=128, fused WMMA kernel for gfx1250.
//
//  gates = [x|h] (B x 256) @ [W_ih^T ; W_hh^T] (256 x 512)  via v_wmma_f32_16x16x32_f16
//  + elementwise LSTM / skip logic + 128-wide dot (new_c_n @ lin_w) fused.
//
// Memory-bound: ~640 MB of HBM traffic vs 68.7 GFLOP -> stream x,h,c once,
// keep W in LDS (f16, 264 KB of 320 KB/WGP), amortized over 256 rows/block.
// W is staged LDS-side by the Tensor Data Mover (one tensor_load_to_lds with
// hardware row padding), tracked with TENSORcnt.

typedef __attribute__((ext_vector_type(16))) _Float16 v16h;
typedef __attribute__((ext_vector_type(8)))  _Float16 v8h;
typedef __attribute__((ext_vector_type(8)))  float    v8f;
typedef __attribute__((ext_vector_type(4)))  unsigned v4u;
typedef __attribute__((ext_vector_type(8)))  int      v8i;
typedef __attribute__((ext_vector_type(4)))  int      v4i;

#define ICn     128
#define HCn     128
#define NGATE   512                 // 4*HC
#define KTOT    256                 // IC + HC
#define WSTR    264                 // padded LDS row stride in halfs (528 B, 16B aligned)
#define TILE_M  16
#define ROWS_PER_BLOCK 256
#define THREADS 256                 // 8 wave32

// dynamic LDS layout:
//   Wlds   : NGATE*WSTR halfs  = 270336 B   (at LDS offset 0)
//   xh     : TILE_M*WSTR halfs =   8448 B
//   rowsum : TILE_M floats     =     64 B
#define LDS_BYTES (NGATE*WSTR*2 + TILE_M*WSTR*2 + TILE_M*4)

// ---------------------------------------------------------------------------
// Prep: W_ih (512x128 f32) and W_hh (512x128 f32) -> packed f16 [512][256]
// (row n = gate index, col k in [0,128) from W_ih, [128,256) from W_hh).
// 256 KB result lives in d_ws; hot in the 192 MB L2 thereafter.
// ---------------------------------------------------------------------------
__global__ void prep_w_kernel(const float* __restrict__ Wih,
                              const float* __restrict__ Whh,
                              unsigned* __restrict__ Wdst)
{
    int idx = blockIdx.x * blockDim.x + threadIdx.x;   // 0 .. 512*128-1 (u32 pairs)
    int n   = idx >> 7;                                // row (gate index)
    int c2  = idx & 127;                               // u32 column
    int k   = c2 * 2;
    float f0, f1;
    if (k < ICn) { f0 = Wih[n * ICn + k];        f1 = Wih[n * ICn + k + 1]; }
    else         { f0 = Whh[n * HCn + (k - ICn)]; f1 = Whh[n * HCn + (k - ICn) + 1]; }
    union { _Float16 h2[2]; unsigned u; } p;
    p.h2[0] = (_Float16)f0;
    p.h2[1] = (_Float16)f1;
    Wdst[idx] = p.u;
}

__device__ __forceinline__ float sigmoidf_(float v) {
    return 1.0f / (1.0f + expf(-v));
}

// ---------------------------------------------------------------------------
// Fused SkipLSTM kernel. Block = 8 waves, 256 rows per block.
// ---------------------------------------------------------------------------
__global__ void __launch_bounds__(THREADS, 1)
skiplstm_kernel(const float* __restrict__ x,  const float* __restrict__ u,
                const float* __restrict__ h,  const float* __restrict__ c,
                const float* __restrict__ delta_u,
                const unsigned char* __restrict__ skip,   // jnp bool_ (1 byte)
                const float* __restrict__ b_ih, const float* __restrict__ b_hh,
                const float* __restrict__ lin_w, const float* __restrict__ lin_b,
                const unsigned* __restrict__ Wf16,        // 512 x 128 u32 (f16 pairs)
                float* __restrict__ out, int B)
{
    extern __shared__ __align__(16) char smem[];
    _Float16* Wlds = (_Float16*)smem;                  // [NGATE][WSTR], LDS offset 0
    _Float16* xh   = Wlds + NGATE * WSTR;              // [TILE_M][WSTR]
    float*    rowsum = (float*)(xh + TILE_M * WSTR);   // [TILE_M]
    unsigned* xhU   = (unsigned*)xh;

    const int tid  = threadIdx.x;
    const int wave = tid >> 5;          // 0..7 : owns hc [16*wave, 16*wave+16)
    const int lane = tid & 31;
    const int half = lane >> 4;         // K-chunk select per WMMA layout
    const int nloc = lane & 15;
    const int hc   = wave * 16 + nloc;

    const long long o_h  = (long long)B;                 // new_h
    const long long o_c  = o_h + (long long)B * HCn;     // new_c
    const long long o_du = o_c + (long long)B * HCn;     // d_u
    const long long o_ns = o_du + (long long)B;          // n_skips_after

    const long long row0 = (long long)blockIdx.x * ROWS_PER_BLOCK;

    // ---- stage W (f16) into LDS once via the Tensor Data Mover ----
    // 2D tensor: 512 rows x 128 dwords, row padding +4 dwords in LDS
    // (pad_interval=6 -> 128 dwords, pad_amount=3 -> 4 dwords => 132-dword
    // stride == WSTR halfs). Padding bytes are never read by the B-fragment
    // loads, so no zero fill is required.
#if __has_builtin(__builtin_amdgcn_tensor_load_to_lds)
    if (tid < 32) {   // one wave issues the DMA (EXEC ignored by TDM)
        const unsigned long long gaddr = (unsigned long long)(const void*)Wf16;
        v4u g0;
        g0[0] = 0x1u;                                  // count=1, user mode
        g0[1] = 0u;                                    // lds_addr = 0 (Wlds)
        g0[2] = (unsigned)(gaddr & 0xFFFFFFFFu);       // global_addr[31:0]
        g0[3] = (unsigned)((gaddr >> 32) & 0x1FFFFFFu) // global_addr[56:32]
              | (2u << 30);                            // type = 2 ("image")
        v8i g1;
        g1[0] = (2 << 16)        // data_size = 4 bytes
              | (1 << 20)        // pad_enable
              | (6 << 22)        // pad_interval: 128 dwords
              | (3 << 25);       // pad_amount:   4 dwords
        g1[1] = (int)(128u << 16);   // tensor_dim0 = 128 (dwords/row)
        g1[2] = (int)(512u << 16);   // tensor_dim1[15:0] = 512 rows
        g1[3] = (int)(128u << 16);   // tile_dim0 = 128
        g1[4] = 512;                 // tile_dim1 = 512
        g1[5] = 128;                 // tensor_dim0_stride = 128 dwords
        g1[6] = 0;
        g1[7] = 0;
        v4i z4 = (v4i)0;
        v8i z8 = (v8i)0;
        // amdgpu-toolchain (clang-23) 6-arg form:
        // (v4u g0, v8i g1, v4i g2, v4i g3, v8i pad, i32 cpol)
        __builtin_amdgcn_tensor_load_to_lds(g0, g1, z4, z4, z8, 0);
        __builtin_amdgcn_s_wait_tensorcnt(0);
    }
#else
    // fallback: cooperative copy with padding
    {
        unsigned* WldsU = (unsigned*)Wlds;
        for (int idx = tid; idx < NGATE * (WSTR / 2); idx += THREADS) {
            int n  = idx / (WSTR / 2);
            int cc = idx - n * (WSTR / 2);
            WldsU[idx] = (cc < KTOT / 2) ? Wf16[n * (KTOT / 2) + cc] : 0u;
        }
    }
#endif

    // per-lane invariants (overlap with the TDM transfer)
    const float linw = lin_w[hc];
    const float bia0 = b_ih[0 * HCn + hc] + b_hh[0 * HCn + hc];
    const float bia1 = b_ih[1 * HCn + hc] + b_hh[1 * HCn + hc];
    const float bia2 = b_ih[2 * HCn + hc] + b_hh[2 * HCn + hc];
    const float bia3 = b_ih[3 * HCn + hc] + b_hh[3 * HCn + hc];

    for (int t = 0; t < ROWS_PER_BLOCK / TILE_M; ++t) {
        const long long tileRow = row0 + (long long)t * TILE_M;

        // ---- stage [x|h] tile as f16 (16 rows x 256 halfs, padded stride) ----
        for (int idx = tid; idx < TILE_M * (KTOT / 2); idx += THREADS) {
            int m  = idx >> 7;        // row in tile
            int c2 = idx & 127;       // u32 column
            int k  = c2 * 2;
            long long grow = tileRow + m;
            float f0, f1;
            if (k < ICn) { f0 = x[grow * ICn + k];         f1 = x[grow * ICn + k + 1]; }
            else         { f0 = h[grow * HCn + (k - ICn)]; f1 = h[grow * HCn + (k - ICn) + 1]; }
            union { _Float16 h2[2]; unsigned uu; } p;
            p.h2[0] = (_Float16)f0;
            p.h2[1] = (_Float16)f1;
            xhU[m * (WSTR / 2) + c2] = p.uu;
        }
        if (tid < TILE_M) rowsum[tid] = 0.0f;
        __syncthreads();

        // ---- WMMA: 16x16 tile per gate, accumulate over K=256 ----
        v8f acc0 = {}, acc1 = {}, acc2 = {}, acc3 = {};
#pragma unroll
        for (int ks = 0; ks < KTOT / 32; ++ks) {
            const int k0 = ks * 32;
            // A fragment: lane holds row nloc, K chunks k0+8*half and +16
            const _Float16* ap = xh + nloc * WSTR + k0 + half * 8;
            v8h alo = *(const v8h*)ap;
            v8h ahi = *(const v8h*)(ap + 16);
            v16h a  = __builtin_shufflevector(alo, ahi,
                        0,1,2,3,4,5,6,7,8,9,10,11,12,13,14,15);

            // B fragments for the 4 gates at column n = gi*128 + hc
#define LOADB(gi, bv)                                                          \
            {                                                                  \
                const _Float16* bp = Wlds + ((gi)*HCn + hc) * WSTR             \
                                          + k0 + half * 8;                     \
                v8h blo = *(const v8h*)bp;                                     \
                v8h bhi = *(const v8h*)(bp + 16);                              \
                bv = __builtin_shufflevector(blo, bhi,                         \
                        0,1,2,3,4,5,6,7,8,9,10,11,12,13,14,15);                \
            }
            v16h b0, b1, b2, b3;
            LOADB(0, b0) LOADB(1, b1) LOADB(2, b2) LOADB(3, b3)
#undef LOADB
            acc0 = __builtin_amdgcn_wmma_f32_16x16x32_f16(false, a, false, b0,
                                                          (short)0, acc0, false, false);
            acc1 = __builtin_amdgcn_wmma_f32_16x16x32_f16(false, a, false, b1,
                                                          (short)0, acc1, false, false);
            acc2 = __builtin_amdgcn_wmma_f32_16x16x32_f16(false, a, false, b2,
                                                          (short)0, acc2, false, false);
            acc3 = __builtin_amdgcn_wmma_f32_16x16x32_f16(false, a, false, b3,
                                                          (short)0, acc3, false, false);
        }

        // ---- elementwise LSTM + skip + dot-product partials ----
#pragma unroll
        for (int r = 0; r < 8; ++r) {
            const int m = r + 8 * half;                  // row within tile
            const long long row = tileRow + m;
            const float uu = u[row];
            const float du = delta_u[row];
            const bool  sk = skip[row] != 0;
            const float bu = rintf(uu);                  // round half-to-even
            const float cc = c[row * HCn + hc];
            const float hh = h[row * HCn + hc];

            const float ig = sigmoidf_(acc0[r] + bia0);
            const float fg = sigmoidf_(acc1[r] + bia1);
            const float gg = tanhf(acc2[r] + bia2);
            const float og = sigmoidf_(acc3[r] + bia3);

            const float nc   = fg * cc + ig * gg;
            const float nh   = og * tanhf(nc);
            const float nc_n = nc * bu;
            const float nh_n = nh * bu;
            const float keep = 1.0f - bu;

            out[o_h + row * HCn + hc] = sk ? hh * keep : nh_n;
            out[o_c + row * HCn + hc] = sk ? cc * keep : nc_n;

            // dot(new_c_n, lin_w): reduce over the 16 hc lanes of this wave
            float contrib = nc_n * linw;
#pragma unroll
            for (int off = 1; off < 16; off <<= 1)
                contrib += __shfl_xor(contrib, off, 32);
            if (nloc == 0) atomicAdd(&rowsum[m], contrib);   // ds_add_f32
        }
        __syncthreads();

        // ---- finalize per-row outputs ----
        if (tid < TILE_M) {
            const int m = tid;
            const long long row = tileRow + m;
            const float uu = u[row];
            const float du = delta_u[row];
            const bool  sk = skip[row] != 0;
            const float bu = rintf(uu);
            const float keep = 1.0f - bu;

            const float dun    = sigmoidf_(rowsum[m] + lin_b[0]);
            const float new_un = dun * bu;
            const float new_us = fminf(fmaxf(uu + du, 0.0f), 1.0f) * keep;

            const float new_u = sk ? new_us : new_un;
            const float d_u   = sk ? du     : dun;
            const float nskp  = ceilf(0.5f / new_u) - 1.0f;

            out[row]        = new_u;
            out[o_du + row] = d_u;
            out[o_ns + row] = nskp;
        }
        __syncthreads();   // protect xh/rowsum before next tile
    }
}

// ---------------------------------------------------------------------------
extern "C" void kernel_launch(void* const* d_in, const int* in_sizes, int n_in,
                              void* d_out, int out_size, void* d_ws, size_t ws_size,
                              hipStream_t stream)
{
    const float* x   = (const float*)d_in[0];
    const float* u   = (const float*)d_in[1];
    const float* h   = (const float*)d_in[2];
    const float* c   = (const float*)d_in[3];
    const float* du  = (const float*)d_in[4];
    const unsigned char* skip = (const unsigned char*)d_in[5];  // jnp bool_
    const float* Wih = (const float*)d_in[6];
    const float* Whh = (const float*)d_in[7];
    const float* bih = (const float*)d_in[8];
    const float* bhh = (const float*)d_in[9];
    const float* lw  = (const float*)d_in[10];
    const float* lb  = (const float*)d_in[11];
    float* out = (float*)d_out;

    unsigned* Wf16 = (unsigned*)d_ws;   // 512*128 u32 = 256 KB of workspace

    const int B = in_sizes[0] / ICn;    // 262144

    // opt in to >64KB dynamic LDS (no-op / harmless if already allowed)
    (void)hipFuncSetAttribute((const void*)skiplstm_kernel,
                              hipFuncAttributeMaxDynamicSharedMemorySize,
                              LDS_BYTES);

    // 1) convert W to f16 workspace (512*128 u32 pairs; 256 blocks x 256 thr)
    prep_w_kernel<<<(NGATE * (KTOT / 2)) / THREADS, THREADS, 0, stream>>>(Wih, Whh, Wf16);

    // 2) fused LSTM: 1024 blocks x 256 threads, 256 rows each
    skiplstm_kernel<<<B / ROWS_PER_BLOCK, THREADS, LDS_BYTES, stream>>>(
        x, u, h, c, du, skip, bih, bhh, lw, lb, Wf16, out, B);
}
